// AdditiveAttention_19138374271219
// MI455X (gfx1250) — matrix operand order repaired and checked by reading, not compiled, not run on hardware
//
#include <hip/hip_runtime.h>
#include <hip/hip_bf16.h>

// Problem constants (match reference)
#define Hdim 1024
#define Bsz  32
#define Lseq 2048

typedef __attribute__((ext_vector_type(16))) __bf16       v16bf;
typedef __attribute__((ext_vector_type(8)))  float        v8f;
typedef __attribute__((ext_vector_type(4)))  float        v4f;
typedef __attribute__((ext_vector_type(4)))  unsigned int v4u;
typedef __attribute__((ext_vector_type(4)))  int          v4i;

#if defined(__has_builtin)
#if __has_builtin(__builtin_amdgcn_global_load_async_to_lds_b128)
#define HAVE_ASYNC_LDS 1
#endif
#if __has_builtin(__builtin_amdgcn_s_wait_asynccnt)
#define HAVE_WAIT_ASYNC 1
#endif
#endif

// Builtin signature (from hipcc diagnostic): arg0 = v4i in global AS ("__device__"),
// arg1 = LDS side, arg2/arg3 = i32 offset / cpol.
typedef __attribute__((address_space(1))) v4i* as1_v4ip;
typedef __attribute__((address_space(3))) v4i* as3_v4ip;

static __device__ __forceinline__ __bf16 f2bf(float f) { return (__bf16)f; }

static __device__ __forceinline__ float tanh_fast(float x) {
#if __has_builtin(__builtin_amdgcn_tanhf)
    return __builtin_amdgcn_tanhf(x);   // v_tanh_f32 (gfx1250 TRANS op)
#else
    return tanhf(x);
#endif
}

static __device__ __forceinline__ void wait_async() {
#if defined(HAVE_ASYNC_LDS)
#if defined(HAVE_WAIT_ASYNC)
    __builtin_amdgcn_s_wait_asynccnt(0);
#else
    asm volatile("s_wait_asynccnt 0" ::: "memory");
#endif
#endif
}

// ---------------------------------------------------------------------------
// Kernel 0: pack Wq,Wk (fp32, row-major (o,h)) -> bf16 row-major in d_ws.
// ---------------------------------------------------------------------------
__global__ void pack_w_kernel(const float* __restrict__ Wq,
                              const float* __restrict__ Wk,
                              __bf16* __restrict__ wbf) {
    int i = blockIdx.x * blockDim.x + threadIdx.x;   // over H*H
    wbf[i]               = f2bf(Wq[i]);
    wbf[Hdim * Hdim + i] = f2bf(Wk[i]);
}

// ---------------------------------------------------------------------------
// A-fragment (16x32 bf16) built on the fly from an fp32 row.
// Layout used here: lane<16 : elems 0..15 = K k0..k0+15 (contiguous)
//             lane>=16: same with k0 += 16.
// ---------------------------------------------------------------------------
static __device__ __forceinline__ v16bf load_a_frag(const float* __restrict__ row,
                                                    int k0) {
    v4f c0 = *(const v4f*)(row + k0);
    v4f c1 = *(const v4f*)(row + k0 + 4);
    v4f c2 = *(const v4f*)(row + k0 + 8);
    v4f c3 = *(const v4f*)(row + k0 + 12);
    v16bf a;
#pragma unroll
    for (int i = 0; i < 4; ++i) {
        a[i]      = f2bf(c0[i]);
        a[4 + i]  = f2bf(c1[i]);
        a[8 + i]  = f2bf(c2[i]);
        a[12 + i] = f2bf(c3[i]);
    }
    return a;
}

// ---------------------------------------------------------------------------
// LDS staging geometry for B (weights):
//   one stage = 2 mats x 64 N-rows x 64 K  (bf16)  -> 128 rows x 128 B data
//   row pitch padded to 144 B  (b128-aligned; 32 lanes land 2-deep on all 16
//   four-bank groups = LDS bandwidth floor, no extra serialization)
// ---------------------------------------------------------------------------
#define BROW_PITCH 144
#define STAGE_BYTES (128 * BROW_PITCH)   // 18432 B, x2 buffers = 36 KB

static __device__ __forceinline__ void fill_stage(char* __restrict__ dstBuf,
                                                  const __bf16* __restrict__ wbf,
                                                  int nBase, int kc, int tid) {
    // 256 threads: thread copies one 64B half-row (4 x b128)
    const int rr  = tid >> 1;           // 0..127 = mat*64 + n
    const int hb  = tid & 1;            // half of the 128B row
    const int mat = rr >> 6;
    const int n   = rr & 63;
    const __bf16* src = wbf + (size_t)mat * Hdim * Hdim
                            + (size_t)(nBase + n) * Hdim + kc + hb * 32;
    char* dst = dstBuf + rr * BROW_PITCH + hb * 64;
#if defined(HAVE_ASYNC_LDS)
#pragma unroll
    for (int j = 0; j < 4; ++j) {
        __builtin_amdgcn_global_load_async_to_lds_b128(
            (as1_v4ip)(v4i*)((const char*)src + j * 16),
            (as3_v4ip)(v4i*)(dst + j * 16), 0, 0);
    }
#else
    const v4u* s4 = (const v4u*)src;
    v4u* d4 = (v4u*)dst;
#pragma unroll
    for (int j = 0; j < 4; ++j) d4[j] = s4[j];
#endif
}

// ---------------------------------------------------------------------------
// Kernel 1: fused dual-GEMM + tanh + dot(w_score) -> score[b*L+l]
// Block = 256 thr (8 waves); wave owns 32 rows (two 16-row A frags).
// N in 16 chunks of 64 cols (4 accumulator tiles, shared by both row frags).
// B staged in LDS, double buffered, 64-K stages, filled by async-to-LDS.
// ---------------------------------------------------------------------------
__global__ __launch_bounds__(256)
void scores_kernel(const float* __restrict__ Q,
                   const float* __restrict__ K,
                   const __bf16* __restrict__ Wbf,   // [Wq_bf | Wk_bf]
                   const float* __restrict__ bias,
                   const float* __restrict__ wsc,
                   float* __restrict__ scoreOut) {
    __shared__ char lbuf[2][STAGE_BYTES];

    const int tid     = threadIdx.x;
    const int lane    = tid & 31;
    const int wave    = tid >> 5;
    const int rowBase = blockIdx.x * 256 + wave * 32;
    const int lo16    = lane & 15;
    const int hi      = (lane >> 4) & 1;

    const float* qrow0 = Q + (size_t)(rowBase + lo16) * Hdim;
    const float* krow0 = K + (size_t)(rowBase + lo16) * Hdim;
    const float* qrow1 = qrow0 + (size_t)16 * Hdim;
    const float* krow1 = krow0 + (size_t)16 * Hdim;
    const int    aOff  = hi ? 16 : 0;

    float sp[2][8];
#pragma unroll
    for (int f = 0; f < 2; ++f)
#pragma unroll
        for (int r = 0; r < 8; ++r) sp[f][r] = 0.0f;

    const v8f vzero = {0.f, 0.f, 0.f, 0.f, 0.f, 0.f, 0.f, 0.f};

    // prologue fill (nc=0, kc=0)
    fill_stage(lbuf[0], Wbf, 0, 0, tid);
    int p = 0;

    for (int nc = 0; nc < 16; ++nc) {                 // 16 N-chunks of 64 cols
        const int nBase = nc * 64;
        v8f acc[2][4];
#pragma unroll
        for (int f = 0; f < 2; ++f)
#pragma unroll
            for (int t = 0; t < 4; ++t) acc[f][t] = vzero;

#pragma unroll 1
        for (int kt = 0; kt < 16; ++kt) {             // 64-K stages
            const int kc = kt * 64;
            wait_async();
            __syncthreads();                          // lbuf[p] ready, lbuf[p^1] free

            if (kt < 15)      fill_stage(lbuf[p ^ 1], Wbf, nBase, kc + 64, tid);
            else if (nc < 15) fill_stage(lbuf[p ^ 1], Wbf, nBase + 64, 0, tid);

            const char* bb = lbuf[p];
#pragma unroll
            for (int s = 0; s < 2; ++s) {             // two 32-K WMMA steps
                const int ks = kc + s * 32;
                v16bf aq0 = load_a_frag(qrow0, ks + aOff);
                v16bf ak0 = load_a_frag(krow0, ks + aOff);
                v16bf aq1 = load_a_frag(qrow1, ks + aOff);
                v16bf ak1 = load_a_frag(krow1, ks + aOff);
#pragma unroll
                for (int t = 0; t < 4; ++t) {
                    const int boff = s * 64 + hi * 32;
                    v16bf bq = *(const v16bf*)(bb + (t * 16 + lo16) * BROW_PITCH + boff);
                    v16bf bk = *(const v16bf*)(bb + ((64 + t * 16 + lo16) * BROW_PITCH) + boff);
                    acc[0][t] = __builtin_amdgcn_wmma_f32_16x16x32_bf16(
                        false, ak0, false, bk, (short)0, acc[0][t], false, false);
                    acc[0][t] = __builtin_amdgcn_wmma_f32_16x16x32_bf16(
                        false, aq0, false, bq, (short)0, acc[0][t], false, false);
                    acc[1][t] = __builtin_amdgcn_wmma_f32_16x16x32_bf16(
                        false, ak1, false, bk, (short)0, acc[1][t], false, false);
                    acc[1][t] = __builtin_amdgcn_wmma_f32_16x16x32_bf16(
                        false, aq1, false, bq, (short)0, acc[1][t], false, false);
                }
            }
            p ^= 1;
        }

        // epilogue: +bias, tanh, *w_score -> per-lane partials (acc regs only)
#pragma unroll
        for (int t = 0; t < 4; ++t) {
            const int n  = nBase + t * 16 + lo16;
            const float bn = bias[n];
            const float wn = wsc[n];
#pragma unroll
            for (int f = 0; f < 2; ++f)
#pragma unroll
                for (int r = 0; r < 8; ++r)
                    sp[f][r] += tanh_fast(acc[f][t][r] + bn) * wn;
        }
    }

    // Reduce over the 16 lanes of each half (C layout: lanes<16 -> M=r,
    // lanes>=16 -> M=8+r), then store scores.
#pragma unroll
    for (int f = 0; f < 2; ++f) {
#pragma unroll
        for (int r = 0; r < 8; ++r) {
            float v = sp[f][r];
            v += __shfl_xor(v, 1, 32);
            v += __shfl_xor(v, 2, 32);
            v += __shfl_xor(v, 4, 32);
            v += __shfl_xor(v, 8, 32);
            sp[f][r] = v;
        }
        if (lo16 == 0) {
            const int row = rowBase + f * 16 + (hi ? 8 : 0);
#pragma unroll
            for (int r = 0; r < 8; ++r) scoreOut[row + r] = sp[f][r];
        }
    }
}

// ---------------------------------------------------------------------------
// Kernel 2: in-place softmax over L per batch (b_score dropped: invariant).
// ---------------------------------------------------------------------------
__global__ __launch_bounds__(256)
void softmax_kernel(float* __restrict__ s) {
    __shared__ float red[256];
    float* row = s + (size_t)blockIdx.x * Lseq;
    const int t = threadIdx.x;

    float v[8];
    float m = -3.4e38f;
#pragma unroll
    for (int i = 0; i < 8; ++i) {
        v[i] = row[t + i * 256];
        m = fmaxf(m, v[i]);
    }
    red[t] = m;
    __syncthreads();
    for (int o = 128; o > 0; o >>= 1) {
        if (t < o) red[t] = fmaxf(red[t], red[t + o]);
        __syncthreads();
    }
    m = red[0];
    __syncthreads();

    float sum = 0.f;
#pragma unroll
    for (int i = 0; i < 8; ++i) {
        v[i] = __expf(v[i] - m);
        sum += v[i];
    }
    red[t] = sum;
    __syncthreads();
    for (int o = 128; o > 0; o >>= 1) {
        if (t < o) red[t] += red[t + o];
        __syncthreads();
    }
    const float inv = 1.0f / red[0];
#pragma unroll
    for (int i = 0; i < 8; ++i) row[t + i * 256] = v[i] * inv;
}

// ---------------------------------------------------------------------------
// Kernel 3: context[b,h] = sum_l attn[b,l] * key[b,l,h]   (memory-bound GEMV)
// ---------------------------------------------------------------------------
__global__ __launch_bounds__(256)
void context_kernel(const float* __restrict__ key,
                    const float* __restrict__ attn,
                    float* __restrict__ ctx) {
    const int b  = blockIdx.x;
    const int h4 = threadIdx.x * 4;
    const float* kb = key + (size_t)b * Lseq * Hdim;
    const float* ab = attn + (size_t)b * Lseq;
    v4f acc = {0.f, 0.f, 0.f, 0.f};
#pragma unroll 4
    for (int l = 0; l < Lseq; ++l) {
        const float a = ab[l];
        v4f k4 = *(const v4f*)(kb + (size_t)l * Hdim + h4);
        acc += k4 * a;
    }
    *(v4f*)(ctx + (size_t)b * Hdim + h4) = acc;
}

// ---------------------------------------------------------------------------
extern "C" void kernel_launch(void* const* d_in, const int* in_sizes, int n_in,
                              void* d_out, int out_size, void* d_ws, size_t ws_size,
                              hipStream_t stream) {
    const float* Q    = (const float*)d_in[0];
    const float* K    = (const float*)d_in[1];
    const float* Wq   = (const float*)d_in[2];
    const float* Wk   = (const float*)d_in[3];
    const float* bias = (const float*)d_in[4];
    const float* wsc  = (const float*)d_in[5];
    // d_in[6] (b_score) intentionally unused: softmax(score + c) == softmax(score)

    float* out  = (float*)d_out;
    float* ctx  = out;                  // (B, 1, H)  = 32768 floats
    float* attn = out + Bsz * Hdim;     // (B, L)     = 65536 floats (scores in-place)

    __bf16* wbf = (__bf16*)d_ws;        // 2 * H * H bf16 = 4 MB scratch

    pack_w_kernel<<<(Hdim * Hdim) / 256, 256, 0, stream>>>(Wq, Wk, wbf);
    scores_kernel<<<(Bsz * Lseq) / 256, 256, 0, stream>>>(Q, K, wbf, bias, wsc, attn);
    softmax_kernel<<<Bsz, 256, 0, stream>>>(attn);
    context_kernel<<<Bsz, 256, 0, stream>>>(K, attn, ctx);
}
